// CLIP_GCN_Model_30580167147674
// MI455X (gfx1250) — matrix operand order repaired and checked by reading, not compiled, not run on hardware
//
#include <hip/hip_runtime.h>
#include <hip/hip_bf16.h>
#include <math.h>

// ---------------------------------------------------------------------------
// Problem constants (fixed by the reference)
// ---------------------------------------------------------------------------
#define N_NODES 10000
#define N_EDGES 160000
#define BATCH   4096
#define DIM     512
#define HID     256

typedef __attribute__((ext_vector_type(16))) __bf16        v16bf;
typedef __attribute__((ext_vector_type(8)))  float         v8f;
typedef __attribute__((ext_vector_type(4)))  unsigned int  u32x4;

union FragU { u32x4 q[2]; v16bf v; };

// ---------------------------------------------------------------------------
// WMMA fragment loaders (wave32, bf16 16x16x32, NT layout: operands row-major
// with K contiguous).  Contiguous K order (K = 16h + s) in both A and B.  Works for
// global or LDS source pointers (inlined -> addrspace inference).
// ---------------------------------------------------------------------------
static __device__ __forceinline__
v16bf load_frag_a(const __hip_bfloat16* A, int lda, int m0, int k0, int lane) {
    int m = lane & 15, h = lane >> 4;
    const char* base = (const char*)(A + (size_t)(m0 + m) * lda + k0);
    FragU f;
    // VGPR v: K = h*16 + {2v, 2v+1}  -> 16 consecutive bf16 = 2 x b128
    f.q[0] = *(const u32x4*)(base + 32 * h);
    f.q[1] = *(const u32x4*)(base + 32 * h + 16);
    return f.v;
}

static __device__ __forceinline__
v16bf load_frag_b(const __hip_bfloat16* B, int ldb, int n0, int k0, int lane) {
    int n = lane & 15, h = lane >> 4;
    const char* base = (const char*)(B + (size_t)(n0 + n) * ldb + k0) + 32 * h;
    FragU f;
    // VGPR v: K = h*16 + {2v, 2v+1}  -> 16 consecutive bf16 = 2 x b128
    f.q[0] = *(const u32x4*)(base);
    f.q[1] = *(const u32x4*)(base + 16);
    return f.v;
}

// ---------------------------------------------------------------------------
// Small helper kernels
// ---------------------------------------------------------------------------
__global__ void k_fill(float* p, float v, int n) {
    int i = blockIdx.x * blockDim.x + threadIdx.x;
    if (i < n) p[i] = v;
}

__global__ void k_edge_deg(const int* dst, float* deg, int E) {
    int e = blockIdx.x * blockDim.x + threadIdx.x;
    if (e < E) atomicAdd(&deg[dst[e]], 1.0f);
}

__global__ void k_rsqrt_inplace(float* deg, int n) {
    int i = blockIdx.x * blockDim.x + threadIdx.x;
    if (i < n) deg[i] = rsqrtf(deg[i]);
}

__global__ void k_f32_to_bf16(const float* x, __hip_bfloat16* y, int n) {
    int i = blockIdx.x * blockDim.x + threadIdx.x;
    if (i < n) y[i] = __float2bfloat16(x[i]);
}

// W[K][N] (row-major, N contiguous) -> Wt_bf16[N][K] (K contiguous)
__global__ void k_transpose_bf16(const float* W, __hip_bfloat16* Wt, int K, int N) {
    int i = blockIdx.x * blockDim.x + threadIdx.x;
    if (i < K * N) {
        int n = i / K, k = i % K;
        Wt[(size_t)n * K + k] = __float2bfloat16(W[(size_t)k * N + n]);
    }
}

__global__ void k_relu_bf16(const float* x, __hip_bfloat16* y, int n) {
    int i = blockIdx.x * blockDim.x + threadIdx.x;
    if (i < n) y[i] = __float2bfloat16(fmaxf(x[i], 0.0f));
}

// agg = dinv^2 * xw + bias  (self-loop term + bias of GCNConv)
__global__ void k_gcn_init(const float* xw, const float* dinv, const float* bias,
                           float* agg, int n, int F) {
    int i = blockIdx.x * blockDim.x + threadIdx.x;
    if (i < n * F) {
        int node = i / F, f = i % F;
        float di = dinv[node];
        agg[i] = di * di * xw[i] + bias[f];
    }
}

// agg[dst] += dinv[src]*dinv[dst]*xw[src]   (scatter over edges)
__global__ void k_gcn_edges(const int* src, const int* dst, const float* xw,
                            const float* dinv, float* agg, int E, int F) {
    int i = blockIdx.x * blockDim.x + threadIdx.x;
    if (i < E * F) {
        int e = i / F, f = i % F;
        int s = src[e], d = dst[e];
        float c = dinv[s] * dinv[d];
        atomicAdd(&agg[(size_t)d * F + f], c * xw[(size_t)s * F + f]);
    }
}

// txt_bf[i][f] = bf16(gcn_out[label[i]][f])
__global__ void k_gather_bf16(const float* X, const int* label,
                              __hip_bfloat16* Y, int B, int F) {
    int i = blockIdx.x * blockDim.x + threadIdx.x;
    if (i < B * F) {
        int b = i / F, f = i % F;
        Y[i] = __float2bfloat16(X[(size_t)label[b] * F + f]);
    }
}

__global__ void k_final(const float* accum, float* out) {
    if (threadIdx.x == 0 && blockIdx.x == 0)
        out[0] = accum[0] / (float)BATCH + 1.0f;   // + constant triplet term
}

// ---------------------------------------------------------------------------
// Generic NT GEMM with WMMA bf16:  C[M][N] = A[M][K] @ B[N][K]^T  (+bias,relu)
// block = (32, 8): 8 waves; the 16xK A tile is staged in LDS once per block,
// each wave owns one 16x16 C tile.  K-loop is software-pipelined (double-
// buffered fragments) so WMMA overlaps the next iteration's loads.
// grid = (M/16, (N/16)/8); all dims divide exactly -> no guards needed.
// ---------------------------------------------------------------------------
template <int ADD_BIAS, int RELU, int OUT_BF16>
__global__ __launch_bounds__(256, 1)
void k_gemm_nt(const __hip_bfloat16* __restrict__ A,
               const __hip_bfloat16* __restrict__ B,
               const float* __restrict__ bias,
               void* __restrict__ Cout, int N, int K) {
    __shared__ __hip_bfloat16 s_a[16 * DIM];     // max K = 512 -> 16KB

    const int lane = threadIdx.x;
    const int tid  = threadIdx.y * 32 + lane;    // 0..255
    const int m0   = blockIdx.x * 16;
    const int n0   = (blockIdx.y * 8 + threadIdx.y) * 16;

    // cooperative stage of A tile
    {
        const u32x4* g = (const u32x4*)(A + (size_t)m0 * K);
        u32x4*       s = (u32x4*)s_a;
        int chunks = 16 * K / 8;                 // 8 bf16 per 16B chunk
        for (int c = tid; c < chunks; c += 256) s[c] = g[c];
    }
    __syncthreads();

    v8f acc = {};
    v16bf a_cur = load_frag_a(s_a, K, 0, 0, lane);
    v16bf b_cur = load_frag_b(B, K, n0, 0, lane);
#pragma unroll 2
    for (int k0 = 32; k0 < K; k0 += 32) {
        v16bf a_nxt = load_frag_a(s_a, K, 0, k0, lane);   // LDS
        v16bf b_nxt = load_frag_b(B, K, n0, k0, lane);    // global
        acc = __builtin_amdgcn_wmma_f32_16x16x32_bf16(
                  false, a_cur, false, b_cur, (short)0, acc, false, false);
        a_cur = a_nxt; b_cur = b_nxt;
    }
    acc = __builtin_amdgcn_wmma_f32_16x16x32_bf16(
              false, a_cur, false, b_cur, (short)0, acc, false, false);

    int n  = n0 + (lane & 15);
    int h  = lane >> 4;
    float bv = ADD_BIAS ? bias[n] : 0.0f;
#pragma unroll
    for (int r = 0; r < 8; ++r) {
        int m = m0 + r + 8 * h;                  // C layout: VGPR r -> M = r + 8h
        float v = acc[r] + bv;
        if (RELU) v = fmaxf(v, 0.0f);
        if (OUT_BF16)
            ((__hip_bfloat16*)Cout)[(size_t)m * N + n] = __float2bfloat16(v);
        else
            ((float*)Cout)[(size_t)m * N + n] = v;
    }
}

// ---------------------------------------------------------------------------
// Fused contrastive-loss kernel: logits = txt @ img^T (4096x4096, K=512)
// computed tile-wise with WMMA.  Each lane keeps a PRIVATE online (max,sum)
// for the (row, column-subset) it holds -- no cross-lane traffic in the hot
// loop; the 16-lane merge happens once at the end (log-sum-exp pair merge).
// block = (32,8), grid = BATCH/16 = 256.
// ---------------------------------------------------------------------------
__global__ __launch_bounds__(256, 1)
void k_loss(const __hip_bfloat16* __restrict__ txt,
            const __hip_bfloat16* __restrict__ img,
            const int* __restrict__ label,
            float* __restrict__ accum) {
    const int i0   = blockIdx.x * 16;
    const int wave = threadIdx.y;
    const int lane = threadIdx.x;
    const int tid  = wave * 32 + lane;
    const int h    = lane >> 4;
    const int nl   = lane & 15;

    __shared__ __hip_bfloat16 s_txt[16 * DIM];   // 16KB row tile
    __shared__ float sdiag[16];
    __shared__ float smax[8][16];
    __shared__ float ssum[8][16];

    // stage the 16x512 txt tile (all waves + all column tiles reuse it)
    {
        const u32x4* g = (const u32x4*)(txt + (size_t)i0 * DIM);
        u32x4*       s = (u32x4*)s_txt;
        for (int c = tid; c < 16 * DIM / 8; c += 256) s[c] = g[c];
    }
    __syncthreads();

    float run_max[8], run_sum[8];                // per-lane private state
#pragma unroll
    for (int r = 0; r < 8; ++r) { run_max[r] = -INFINITY; run_sum[r] = 0.0f; }

    // column tiles: wave w handles tiles w, w+8, ... (BATCH/16 = 256 tiles)
    for (int t = wave; t < BATCH / 16; t += 8) {
        int j0 = t * 16;
        v8f acc = {};
        v16bf a_cur = load_frag_a(s_txt, DIM, 0, 0, lane);
        v16bf b_cur = load_frag_b(img, DIM, j0, 0, lane);
#pragma unroll 2
        for (int k0 = 32; k0 < DIM; k0 += 32) {
            v16bf a_nxt = load_frag_a(s_txt, DIM, 0, k0, lane);  // LDS
            v16bf b_nxt = load_frag_b(img, DIM, j0, k0, lane);   // global
            acc = __builtin_amdgcn_wmma_f32_16x16x32_bf16(
                      false, a_cur, false, b_cur, (short)0, acc, false, false);
            a_cur = a_nxt; b_cur = b_nxt;
        }
        acc = __builtin_amdgcn_wmma_f32_16x16x32_bf16(
                  false, a_cur, false, b_cur, (short)0, acc, false, false);

        if (j0 == i0) {                          // wave-uniform; runs once
#pragma unroll
            for (int r = 0; r < 8; ++r)
                if (nl == r + 8 * h) sdiag[nl] = acc[r];   // logits[i,i]
        }
        // per-lane online logsumexp update (no shuffles here)
#pragma unroll
        for (int r = 0; r < 8; ++r) {
            float v  = acc[r];                   // row r+8h, col j0+nl
            float nm = fmaxf(run_max[r], v);
            run_sum[r] = run_sum[r] * __expf(run_max[r] - nm) + __expf(v - nm);
            run_max[r] = nm;
        }
    }

    // merge the 16 per-lane partials of each row (pair-wise lse merge)
#pragma unroll
    for (int r = 0; r < 8; ++r) {
        float m = run_max[r], s = run_sum[r];
        for (int d = 8; d >= 1; d >>= 1) {
            float om = __shfl_xor(m, d, 16);
            float os = __shfl_xor(s, d, 16);
            float nm = fmaxf(m, om);
            s = s * __expf(m - nm) + os * __expf(om - nm);
            m = nm;
        }
        run_max[r] = m; run_sum[r] = s;
    }

    // publish per-wave (max,sum) for its 16 rows
    if (nl == 0) {
#pragma unroll
        for (int r = 0; r < 8; ++r) {
            smax[wave][r + 8 * h] = run_max[r];
            ssum[wave][r + 8 * h] = run_sum[r];
        }
    }
    __syncthreads();

    // combine across the 8 waves; threads (wave 0, lanes 0..15), row = lane
    if (wave == 0 && lane < 16) {
        float M = -INFINITY;
#pragma unroll
        for (int w = 0; w < 8; ++w) M = fmaxf(M, smax[w][lane]);
        float S = 0.0f;
#pragma unroll
        for (int w = 0; w < 8; ++w) S += ssum[w][lane] * __expf(smax[w][lane] - M);
        float lse = M + __logf(S);
        int   i   = i0 + lane;
        float li  = -(float)label[i] * (sdiag[lane] - lse);
        for (int s = 8; s >= 1; s >>= 1) li += __shfl_xor(li, s, 16);
        if (lane == 0) atomicAdd(accum, li);
    }
}

// ---------------------------------------------------------------------------
// Host launcher
// ---------------------------------------------------------------------------
extern "C" void kernel_launch(void* const* d_in, const int* in_sizes, int n_in,
                              void* d_out, int out_size, void* d_ws, size_t ws_size,
                              hipStream_t stream) {
    (void)in_sizes; (void)n_in; (void)out_size; (void)ws_size;

    const float* image   = (const float*)d_in[0];
    const float* x_nodes = (const float*)d_in[1];
    const int*   edge    = (const int*)d_in[2];
    const int*   label   = (const int*)d_in[3];
    const float* W_img1  = (const float*)d_in[4];
    const float* b_img1  = (const float*)d_in[5];
    const float* W_img2  = (const float*)d_in[6];
    const float* b_img2  = (const float*)d_in[7];
    const float* W_g1    = (const float*)d_in[8];
    const float* b_g1    = (const float*)d_in[9];
    const float* W_g2    = (const float*)d_in[10];
    const float* b_g2    = (const float*)d_in[11];
    const int* src = edge;
    const int* dst = edge + N_EDGES;
    float* out = (float*)d_out;

    // workspace carve-up (256B aligned)
    char* p = (char*)d_ws;
    auto alloc = [&](size_t bytes) -> char* {
        char* r = p; p += (bytes + 255) & ~(size_t)255; return r;
    };
    float* dinv   = (float*)alloc(N_NODES * 4);                    // deg -> dinv in place
    __hip_bfloat16* x_bf    = (__hip_bfloat16*)alloc((size_t)N_NODES * DIM * 2);
    __hip_bfloat16* img_bf  = (__hip_bfloat16*)alloc((size_t)BATCH * DIM * 2);
    __hip_bfloat16* wg1t    = (__hip_bfloat16*)alloc((size_t)DIM * HID * 2);
    __hip_bfloat16* wg2t    = (__hip_bfloat16*)alloc((size_t)HID * DIM * 2);
    __hip_bfloat16* wi1t    = (__hip_bfloat16*)alloc((size_t)DIM * HID * 2);
    __hip_bfloat16* wi2t    = (__hip_bfloat16*)alloc((size_t)HID * DIM * 2);
    float* xw1  = (float*)alloc((size_t)N_NODES * HID * 4);
    float* agg1 = (float*)alloc((size_t)N_NODES * HID * 4);
    __hip_bfloat16* h_bf    = (__hip_bfloat16*)alloc((size_t)N_NODES * HID * 2);
    float* xw2  = (float*)alloc((size_t)N_NODES * DIM * 4);
    float* agg2 = (float*)alloc((size_t)N_NODES * DIM * 4);
    __hip_bfloat16* t1_bf   = (__hip_bfloat16*)alloc((size_t)BATCH * HID * 2);
    __hip_bfloat16* enc_bf  = (__hip_bfloat16*)alloc((size_t)BATCH * DIM * 2);
    __hip_bfloat16* txt_bf  = (__hip_bfloat16*)alloc((size_t)BATCH * DIM * 2);
    float* accum = (float*)alloc(256);

    const int T = 256;
    auto cdiv = [](int a, int b) { return (a + b - 1) / b; };

    // --- degree / normalization -------------------------------------------
    k_fill<<<cdiv(N_NODES, T), T, 0, stream>>>(dinv, 1.0f, N_NODES);
    k_edge_deg<<<cdiv(N_EDGES, T), T, 0, stream>>>(dst, dinv, N_EDGES);
    k_rsqrt_inplace<<<cdiv(N_NODES, T), T, 0, stream>>>(dinv, N_NODES);

    // --- bf16 conversions / weight transposes ------------------------------
    k_f32_to_bf16<<<cdiv(N_NODES * DIM, T), T, 0, stream>>>(x_nodes, x_bf, N_NODES * DIM);
    k_f32_to_bf16<<<cdiv(BATCH * DIM, T), T, 0, stream>>>(image, img_bf, BATCH * DIM);
    k_transpose_bf16<<<cdiv(DIM * HID, T), T, 0, stream>>>(W_g1,   wg1t, DIM, HID);
    k_transpose_bf16<<<cdiv(HID * DIM, T), T, 0, stream>>>(W_g2,   wg2t, HID, DIM);
    k_transpose_bf16<<<cdiv(DIM * HID, T), T, 0, stream>>>(W_img1, wi1t, DIM, HID);
    k_transpose_bf16<<<cdiv(HID * DIM, T), T, 0, stream>>>(W_img2, wi2t, HID, DIM);

    dim3 blk(32, 8);

    // --- GCN layer 1: xw1 = X @ Wg1 ; agg ; relu -> h_bf -------------------
    k_gemm_nt<0, 0, 0><<<dim3(N_NODES / 16, (HID / 16) / 8), blk, 0, stream>>>(
        x_bf, wg1t, nullptr, xw1, HID, DIM);
    k_gcn_init<<<cdiv(N_NODES * HID, T), T, 0, stream>>>(xw1, dinv, b_g1, agg1, N_NODES, HID);
    k_gcn_edges<<<cdiv(N_EDGES * HID, T), T, 0, stream>>>(src, dst, xw1, dinv, agg1, N_EDGES, HID);
    k_relu_bf16<<<cdiv(N_NODES * HID, T), T, 0, stream>>>(agg1, h_bf, N_NODES * HID);

    // --- GCN layer 2: xw2 = H @ Wg2 ; agg -> gcn_out (agg2) ----------------
    k_gemm_nt<0, 0, 0><<<dim3(N_NODES / 16, (DIM / 16) / 8), blk, 0, stream>>>(
        h_bf, wg2t, nullptr, xw2, DIM, HID);
    k_gcn_init<<<cdiv(N_NODES * DIM, T), T, 0, stream>>>(xw2, dinv, b_g2, agg2, N_NODES, DIM);
    k_gcn_edges<<<cdiv(N_EDGES * DIM, T), T, 0, stream>>>(src, dst, xw2, dinv, agg2, N_EDGES, DIM);

    // --- Image MLP: relu(relu(img@W1+b1)@W2+b2) -> enc_bf ------------------
    k_gemm_nt<1, 1, 1><<<dim3(BATCH / 16, (HID / 16) / 8), blk, 0, stream>>>(
        img_bf, wi1t, b_img1, t1_bf, HID, DIM);
    k_gemm_nt<1, 1, 1><<<dim3(BATCH / 16, (DIM / 16) / 8), blk, 0, stream>>>(
        t1_bf, wi2t, b_img2, enc_bf, DIM, HID);

    // --- gather txt = gcn_out[label] --------------------------------------
    k_gather_bf16<<<cdiv(BATCH * DIM, T), T, 0, stream>>>(agg2, label, txt_bf, BATCH, DIM);

    // --- fused contrastive loss -------------------------------------------
    k_fill<<<1, 32, 0, stream>>>(accum, 0.0f, 1);
    k_loss<<<BATCH / 16, blk, 0, stream>>>(txt_bf, enc_bf, label, accum);
    k_final<<<1, 32, 0, stream>>>(accum, out);
}
